// SceneGraphEncoder_20220706030367
// MI455X (gfx1250) — matrix operand order repaired and checked by reading, not compiled, not run on hardware
//
#include <hip/hip_runtime.h>
#include <hip/hip_bf16.h>

typedef __attribute__((ext_vector_type(16))) _Float16 v16h;
typedef __attribute__((ext_vector_type(8)))  _Float16 v8h;
typedef __attribute__((ext_vector_type(2)))  _Float16 v2h;
typedef __attribute__((ext_vector_type(8)))  float    v8f;
typedef __attribute__((ext_vector_type(4)))  float    v4f;

// ---------------------------------------------------------------------------
// Tiled GEMM: C[M,Nn] = act(A[M,K] @ W[K,Nn] + bias)
// f32 -> f16 staged through LDS, f32 accumulate via v_wmma_f32_16x16x32_f16.
// Block = 128 threads = 4 waves; block tile 32(M) x 64(N); wave tile 16x32.
// LDS layouts chosen so each WMMA fragment is contiguous 16B chunks:
//   A row-major [m][k]  : lane frag = A[m][8*lh+0..7] ++ A[m][16+8*lh+0..7]
//   B transposed [n][k] : lane frag = Bt[n][8*lh+0..7] ++ Bt[n][16+8*lh+0..7]
// B staging packs k-pairs with v_cvt_pk_f16_f32 -> ds_store_b32.
// ---------------------------------------------------------------------------
__global__ __launch_bounds__(128)
void gemm_wmma_kernel(const float* __restrict__ A, const float* __restrict__ W,
                      const float* __restrict__ bias, float* __restrict__ C,
                      int M, int K, int Nn, int act)
{
    __shared__ __align__(16) _Float16 As[32][40];   // [m][k], 80B row stride
    __shared__ __align__(16) _Float16 Bt[64][40];   // [n][k], transposed

    const int blockM = blockIdx.x * 32;
    const int blockN = blockIdx.y * 64;

    const int tid  = threadIdx.x;
    const int lane = tid & 31;
    const int wid  = tid >> 5;
    const int lh   = lane >> 4;          // half-wave
    const int lr   = lane & 15;
    const int mi   = (wid >> 1) << 4;    // 0 / 16
    const int nj   = (wid & 1) << 5;     // 0 / 32

    const bool interiorMN = (blockM + 32 <= M) && (blockN + 64 <= Nn);

    v8f acc0 = {}, acc1 = {};

    for (int k0 = 0; k0 < K; k0 += 32) {
        if (interiorMN && (k0 + 32 <= K)) {
            // ---- A tile 32x32: 8 contiguous halfs per thread ----
            {
                int r  = tid >> 2;
                int kc = (tid & 3) << 3;
                const float* src = &A[(size_t)(blockM + r) * K + (k0 + kc)];
                v4f f0 = *(const v4f*)(src);
                v4f f1 = *(const v4f*)(src + 4);
                v8h h;
#pragma unroll
                for (int i = 0; i < 4; ++i) { h[i] = (_Float16)f0[i]; h[4 + i] = (_Float16)f1[i]; }
                *(v8h*)&As[r][kc] = h;
            }
            // ---- B tile 32x64 -> Bt[n][k]: one k-pair x 8 n per thread ----
            {
                int kp = tid >> 3;                   // k-pair 0..15 -> k = 2kp, 2kp+1
                int nc = (tid & 7) << 3;             // n col, step 8
                const float* s0 = &W[(size_t)(k0 + 2 * kp) * Nn + (blockN + nc)];
                const float* s1 = s0 + Nn;
                v4f a0 = *(const v4f*)(s0);
                v4f a1 = *(const v4f*)(s0 + 4);
                v4f b0 = *(const v4f*)(s1);
                v4f b1 = *(const v4f*)(s1 + 4);
#pragma unroll
                for (int i = 0; i < 4; ++i) {
                    v2h p0; p0[0] = (_Float16)a0[i]; p0[1] = (_Float16)b0[i];
                    *(v2h*)&Bt[nc + i][2 * kp] = p0;
                    v2h p1; p1[0] = (_Float16)a1[i]; p1[1] = (_Float16)b1[i];
                    *(v2h*)&Bt[nc + 4 + i][2 * kp] = p1;
                }
            }
        } else {
            for (int t = tid; t < 32 * 32; t += 128) {
                int r = t >> 5, c = t & 31;
                int gr = blockM + r, gc = k0 + c;
                As[r][c] = (gr < M && gc < K) ? (_Float16)A[(size_t)gr * K + gc] : (_Float16)0.f;
            }
            for (int t = tid; t < 32 * 64; t += 128) {
                int r = t >> 6, c = t & 63;          // r: k, c: n
                int kr = k0 + r, nc = blockN + c;
                Bt[c][r] = (kr < K && nc < Nn) ? (_Float16)W[(size_t)kr * Nn + nc] : (_Float16)0.f;
            }
        }
        __syncthreads();

        // ---- fragments: contiguous 16B LDS loads ----
        v8h a0 = *(const v8h*)&As[mi + lr][8 * lh];
        v8h a1 = *(const v8h*)&As[mi + lr][16 + 8 * lh];
        v16h a = __builtin_shufflevector(a0, a1,
                 0, 1, 2, 3, 4, 5, 6, 7, 8, 9, 10, 11, 12, 13, 14, 15);

        v8h b00 = *(const v8h*)&Bt[nj + lr][8 * lh];
        v8h b01 = *(const v8h*)&Bt[nj + lr][16 + 8 * lh];
        v16h b0 = __builtin_shufflevector(b00, b01,
                  0, 1, 2, 3, 4, 5, 6, 7, 8, 9, 10, 11, 12, 13, 14, 15);
        acc0 = __builtin_amdgcn_wmma_f32_16x16x32_f16(false, a, false, b0,
                                                      (short)0, acc0, false, false);

        v8h b10 = *(const v8h*)&Bt[nj + 16 + lr][8 * lh];
        v8h b11 = *(const v8h*)&Bt[nj + 16 + lr][16 + 8 * lh];
        v16h b1 = __builtin_shufflevector(b10, b11,
                  0, 1, 2, 3, 4, 5, 6, 7, 8, 9, 10, 11, 12, 13, 14, 15);
        acc1 = __builtin_amdgcn_wmma_f32_16x16x32_f16(false, a, false, b1,
                                                      (short)0, acc1, false, false);

        __syncthreads();
    }

    // ---- store: C/D layout -> M = r + 8*lh, N = lr ----
#pragma unroll
    for (int r = 0; r < 8; ++r) {
        int row  = blockM + mi + r + 8 * lh;
        int col0 = blockN + nj + lr;
        int col1 = col0 + 16;
        if (row < M) {
            if (col0 < Nn) {
                float v = acc0[r] + (bias ? bias[col0] : 0.f);
                if (act == 1) v = fmaxf(v, 0.f);
                C[(size_t)row * Nn + col0] = v;
            }
            if (col1 < Nn) {
                float v = acc1[r] + (bias ? bias[col1] : 0.f);
                if (act == 1) v = fmaxf(v, 0.f);
                C[(size_t)row * Nn + col1] = v;
            }
        }
    }
}

// ---------------------------------------------------------------------------
// Atomic float max via signed-max / unsigned-min trick (init must be -inf)
// ---------------------------------------------------------------------------
__device__ __forceinline__ void atomicMaxFloat(float* addr, float value)
{
    if (value >= 0.f)
        atomicMax((int*)addr, __float_as_int(value));
    else
        atomicMin((unsigned int*)addr, (unsigned int)__float_as_int(value));
}

// ---------------------------------------------------------------------------
// GAT segment-softmax kernels
// ---------------------------------------------------------------------------
__global__ __launch_bounds__(256)
void init_gat_kernel(float* __restrict__ gout, float* __restrict__ mx,
                     float* __restrict__ dn, int N, int Kh, int KC)
{
    int i = blockIdx.x * blockDim.x + threadIdx.x;
    if (i < N * KC) gout[i] = 0.f;
    if (i < N * Kh) { mx[i] = -__builtin_inff(); dn[i] = 0.f; }
}

__global__ __launch_bounds__(256)
void edge_logit_kernel(const float* __restrict__ xl, const float* __restrict__ xr,
                       const float* __restrict__ ef, const float* __restrict__ att,
                       const int* __restrict__ src, const int* __restrict__ dst,
                       float* __restrict__ logit, float* __restrict__ mx,
                       int E, int Kh, int C)
{
    int idx = blockIdx.x * blockDim.x + threadIdx.x;
    if (idx >= E * Kh) return;
    int e = idx / Kh, k = idx - e * Kh;
    int s = src[e], d = dst[e];
    int KC = Kh * C;
    const v4f* pl = (const v4f*)(xl + (size_t)s * KC + k * C);
    const v4f* pr = (const v4f*)(xr + (size_t)d * KC + k * C);
    const v4f* pe = (const v4f*)(ef + (size_t)e * KC + k * C);
    const v4f* pa = (const v4f*)(att + k * C);
    float acc = 0.f;
    for (int c4 = 0; c4 < (C >> 2); ++c4) {
        v4f vl = pl[c4], vr = pr[c4], ve = pe[c4], va = pa[c4];
#pragma unroll
        for (int i = 0; i < 4; ++i) {
            float m = vl[i] + vr[i] + ve[i];
            m = m > 0.f ? m : 0.2f * m;                // leaky_relu(0.2)
            acc += m * va[i];
        }
    }
    logit[(size_t)e * Kh + k] = acc;
    atomicMaxFloat(&mx[(size_t)d * Kh + k], acc);
}

__global__ __launch_bounds__(256)
void edge_exp_kernel(float* __restrict__ logit, const float* __restrict__ mx,
                     float* __restrict__ dn, const int* __restrict__ dst,
                     int E, int Kh)
{
    int idx = blockIdx.x * blockDim.x + threadIdx.x;
    if (idx >= E * Kh) return;
    int e = idx / Kh, k = idx - e * Kh;
    int d = dst[e];
    float ex = expf(logit[idx] - mx[(size_t)d * Kh + k]);
    logit[idx] = ex;
    atomicAdd(&dn[(size_t)d * Kh + k], ex);
}

// 4 features per thread: vector load + 4 atomic adds
__global__ __launch_bounds__(256)
void edge_scatter_kernel(const float* __restrict__ xl, const float* __restrict__ ex,
                         const float* __restrict__ dn,
                         const int* __restrict__ src, const int* __restrict__ dst,
                         float* __restrict__ gout, int E, int Kh, int C)
{
    int KC4 = (Kh * C) >> 2;
    long idx = (long)blockIdx.x * blockDim.x + threadIdx.x;
    if (idx >= (long)E * KC4) return;
    int e  = (int)(idx / KC4);
    int j4 = (int)(idx - (long)e * KC4);       // 4-feature chunk index
    int k  = (j4 << 2) / C;
    int s = src[e], d = dst[e];
    float alpha = ex[(size_t)e * Kh + k] / fmaxf(dn[(size_t)d * Kh + k], 1e-16f);
    v4f v = *(const v4f*)(xl + (size_t)s * (KC4 << 2) + (j4 << 2));
    float* g = gout + (size_t)d * (KC4 << 2) + (j4 << 2);
#pragma unroll
    for (int i = 0; i < 4; ++i) atomicAdd(&g[i], v[i] * alpha);
}

// ---------------------------------------------------------------------------
// LayerNorm + residual + ELU (+ post-add), one wave32 per row
// out = elu( LN(gin + bias)*g + b (+res_in) ) (+res_post)
// ---------------------------------------------------------------------------
__global__ __launch_bounds__(256)
void ln_post_kernel(const float* __restrict__ gin, const float* __restrict__ bias,
                    const float* __restrict__ ln_g, const float* __restrict__ ln_b,
                    const float* __restrict__ res_in, const float* __restrict__ res_post,
                    float* __restrict__ out, int rows, int D)
{
    int wave = (int)((blockIdx.x * (long)blockDim.x + threadIdx.x) >> 5);
    int lane = threadIdx.x & 31;
    if (wave >= rows) return;
    int nch = D >> 5;                 // 4 or 8
    float vals[8];
    float s = 0.f, s2 = 0.f;
    for (int i = 0; i < nch; ++i) {
        int c = lane + (i << 5);
        float v = gin[(size_t)wave * D + c] + bias[c];
        vals[i] = v;
        s += v; s2 += v * v;
    }
    for (int off = 16; off > 0; off >>= 1) {
        s  += __shfl_xor(s, off, 32);
        s2 += __shfl_xor(s2, off, 32);
    }
    float mu  = s / (float)D;
    float var = s2 / (float)D - mu * mu;
    float inv = rsqrtf(var + 1e-5f);
    for (int i = 0; i < nch; ++i) {
        int c = lane + (i << 5);
        float v = (vals[i] - mu) * inv * ln_g[c] + ln_b[c];
        if (res_in) v += res_in[(size_t)wave * D + c];
        v = v > 0.f ? v : (expf(v) - 1.f);          // elu
        if (res_post) v += res_post[(size_t)wave * D + c];
        out[(size_t)wave * D + c] = v;
    }
}

// ---------------------------------------------------------------------------
// Attentional pooling kernels
// ---------------------------------------------------------------------------
__global__ __launch_bounds__(1024)
void init_pool_kernel(float* gmax, float* gden, float* gpool, int B)
{
    int t = threadIdx.x;
    if (t < B) { gmax[t] = -__builtin_inff(); gden[t] = 0.f; }
    if (t < B * 128) gpool[t] = 0.f;
}

__global__ __launch_bounds__(256)
void gate_logit_kernel(const float* __restrict__ gh, const float* __restrict__ w2,
                       const float* __restrict__ b2, const int* __restrict__ batch,
                       float* __restrict__ gl, float* __restrict__ gmax, int N)
{
    int i = blockIdx.x * blockDim.x + threadIdx.x;
    if (i >= N) return;
    float a = b2[0];
    const v4f* p = (const v4f*)(gh + (size_t)i * 64);
    const v4f* w = (const v4f*)w2;
    for (int c = 0; c < 16; ++c) {
        v4f vp = p[c], vw = w[c];
#pragma unroll
        for (int j = 0; j < 4; ++j) a += vp[j] * vw[j];
    }
    gl[i] = a;
    atomicMaxFloat(&gmax[batch[i]], a);
}

__global__ __launch_bounds__(256)
void pool_exp_kernel(float* __restrict__ gl, const float* __restrict__ gmax,
                     float* __restrict__ gden, const int* __restrict__ batch, int N)
{
    int i = blockIdx.x * blockDim.x + threadIdx.x;
    if (i >= N) return;
    int b = batch[i];
    float ge = expf(gl[i] - gmax[b]);
    gl[i] = ge;
    atomicAdd(&gden[b], ge);
}

__global__ __launch_bounds__(256)
void pool_acc_kernel(const float* __restrict__ h, const float* __restrict__ gl,
                     const float* __restrict__ gden, const int* __restrict__ batch,
                     float* __restrict__ gpool, int N)
{
    long idx = (long)blockIdx.x * blockDim.x + threadIdx.x;   // N * 32 chunks of 4
    if (idx >= (long)N * 32) return;
    int i  = (int)(idx >> 5);
    int c4 = (int)(idx & 31);
    int b = batch[i];
    float w = gl[i] / gden[b];
    v4f v = *(const v4f*)(h + (size_t)i * 128 + (c4 << 2));
    float* g = gpool + b * 128 + (c4 << 2);
#pragma unroll
    for (int j = 0; j < 4; ++j) atomicAdd(&g[j], v[j] * w);
}

__global__ __launch_bounds__(256)
void scene_mlp_kernel(const float* __restrict__ gpool,
                      const float* __restrict__ w1, const float* __restrict__ b1,
                      const float* __restrict__ w2, const float* __restrict__ b2,
                      float* __restrict__ z)
{
    __shared__ float hid[8][64];
    int t = threadIdx.x;
    for (int idx = t; idx < 8 * 64; idx += 256) {
        int bb = idx >> 6, j = idx & 63;
        float a = b1[j];
        for (int c = 0; c < 128; ++c) a += gpool[bb * 128 + c] * w1[c * 64 + j];
        hid[bb][j] = fmaxf(a, 0.f);
    }
    __syncthreads();
    if (t < 8 * 32) {
        int bb = t >> 5, o = t & 31;
        float a = b2[o];
        for (int j = 0; j < 64; ++j) a += hid[bb][j] * w2[j * 32 + o];
        z[bb * 32 + o] = a;
    }
}

// ---------------------------------------------------------------------------
// Host launcher
// ---------------------------------------------------------------------------
extern "C" void kernel_launch(void* const* d_in, const int* in_sizes, int n_in,
                              void* d_out, int out_size, void* d_ws, size_t ws_size,
                              hipStream_t stream)
{
    const float* x     = (const float*)d_in[0];
    const int*   ei    = (const int*)  d_in[1];
    const float* ea    = (const float*)d_in[2];
    const int*   batch = (const int*)  d_in[3];

    const float* P[64];
    for (int i = 4; i < n_in && i < 68; ++i) P[i - 4] = (const float*)d_in[i];

    const int N = in_sizes[3];          // 50000
    const int E = in_sizes[1] / 2;      // 400000
    const int B = 8;
    const int* src = ei;
    const int* dst = ei + E;

    // ---- workspace carve-out (floats) ----
    float* ws = (float*)d_ws;
    size_t off = 0;
    float* h_skip = ws + off; off += (size_t)N * 128;
    float* xl     = ws + off; off += (size_t)N * 256;
    float* xr     = ws + off; off += (size_t)N * 256;
    float* ef     = ws + off; off += (size_t)E * 256;
    float* lg     = ws + off; off += (size_t)E * 4;
    float* mx     = ws + off; off += (size_t)N * 4;
    float* dn     = ws + off; off += (size_t)N * 4;
    float* go     = ws + off; off += (size_t)N * 256;
    float* h1     = ws + off; off += (size_t)N * 256;
    float* h2     = ws + off; off += (size_t)N * 256;
    float* rp     = ws + off; off += (size_t)N * 256;
    float* rp2    = ws + off; off += (size_t)N * 128;
    float* gh     = ws + off; off += (size_t)N * 64;
    float* gl     = ws + off; off += (size_t)N;
    float* gmax   = ws + off; off += B;
    float* gden   = ws + off; off += B;
    float* gpool  = ws + off; off += (size_t)B * 128;
    (void)ws_size;

    // ---- output layout: z[B,32] | node_emb[N,128] | edge_repr[E,128] ----
    float* z_out = (float*)d_out;
    float* nemb  = z_out + B * 32;
    float* eout  = nemb + (size_t)N * 128;
    (void)out_size;

    auto gemm = [&](const float* A, const float* W, const float* bias, float* C,
                    int M, int K, int Nn, int act) {
        dim3 grid((M + 31) / 32, (Nn + 63) / 64);
        gemm_wmma_kernel<<<grid, dim3(128), 0, stream>>>(A, W, bias, C, M, K, Nn, act);
    };

    auto gat = [&](const float* hin, int fi, int Kh, int C, int pw) {
        int KC = Kh * C;
        gemm(hin, P[pw + 0], P[pw + 1], xl, N, fi, KC, 0);   // wl, bl
        gemm(hin, P[pw + 2], P[pw + 3], xr, N, fi, KC, 0);   // wr, br
        gemm(ea,  P[pw + 4], nullptr,   ef, E, 13, KC, 0);   // we
        int tot = N * KC;
        init_gat_kernel<<<(tot + 255) / 256, 256, 0, stream>>>(go, mx, dn, N, Kh, KC);
        int ek = E * Kh;
        edge_logit_kernel<<<(ek + 255) / 256, 256, 0, stream>>>(
            xl, xr, ef, P[pw + 5], src, dst, lg, mx, E, Kh, C);
        edge_exp_kernel<<<(ek + 255) / 256, 256, 0, stream>>>(lg, mx, dn, dst, E, Kh);
        long ejc = (long)E * (KC >> 2);
        edge_scatter_kernel<<<(unsigned)((ejc + 255) / 256), 256, 0, stream>>>(
            xl, lg, dn, src, dst, go, E, Kh, C);
    };

    // ---- node encoder: h_skip = mlp2(x) ----
    gemm(x,  P[0], P[1], xl,     N, 10,  256, 1);
    gemm(xl, P[2], P[3], h_skip, N, 256, 128, 0);

    // ---- GAT layer 1 (128 -> 4x64) ----
    gat(h_skip, 128, 4, 64, 8);
    ln_post_kernel<<<(N + 7) / 8, 256, 0, stream>>>(
        go, P[14], P[29], P[30], nullptr, nullptr, h1, N, 256);

    // ---- GAT layer 2 (256 -> 4x64) + residual + skip projection ----
    gat(h1, 256, 4, 64, 15);
    gemm(h_skip, P[35], P[36], rp, N, 128, 256, 0);          // rp_w, rp_b
    ln_post_kernel<<<(N + 7) / 8, 256, 0, stream>>>(
        go, P[21], P[31], P[32], h1, rp, h2, N, 256);

    // ---- GAT layer 3 (256 -> 2x64) + projected residual ----
    gat(h2, 256, 2, 64, 22);
    gemm(h2, P[37], P[38], rp2, N, 256, 128, 0);             // rp2_w, rp2_b
    ln_post_kernel<<<(N + 7) / 8, 256, 0, stream>>>(
        go, P[28], P[33], P[34], rp2, nullptr, nemb, N, 128);

    // ---- edge encoder: edge_repr = mlp2(edge_attr) ----
    gemm(ea, P[4], P[5], ef,   E, 13, 64, 1);
    gemm(ef, P[6], P[7], eout, E, 64, 128, 0);

    // ---- attentional pooling + scene MLP ----
    gemm(nemb, P[39], P[40], gh, N, 128, 64, 1);             // gate_w1 relu
    init_pool_kernel<<<1, 1024, 0, stream>>>(gmax, gden, gpool, B);
    gate_logit_kernel<<<(N + 255) / 256, 256, 0, stream>>>(gh, P[41], P[42], batch, gl, gmax, N);
    pool_exp_kernel<<<(N + 255) / 256, 256, 0, stream>>>(gl, gmax, gden, batch, N);
    pool_acc_kernel<<<(unsigned)(((long)N * 32 + 255) / 256), 256, 0, stream>>>(
        nemb, gl, gden, batch, gpool, N);
    scene_mlp_kernel<<<1, 256, 0, stream>>>(gpool, P[43], P[44], P[45], P[46], z_out);
}
